// Clustering_20753281974744
// MI455X (gfx1250) — compile-verified, matrix operand order checked by reading, not run on hardware
//
#include <hip/hip_runtime.h>
#include <hip/hip_bf16.h>
#include <math.h>

typedef __bf16 bf16;
typedef __attribute__((ext_vector_type(16))) __bf16 v16bf;
typedef __attribute__((ext_vector_type(8)))  __bf16 v8bf;
typedef __attribute__((ext_vector_type(8)))  float  v8f;

// ---------- helpers ----------
static __device__ __forceinline__ v8f zero_v8f() {
    v8f z = {0.f,0.f,0.f,0.f,0.f,0.f,0.f,0.f};
    return z;
}

// Load a 16x32 bf16 WMMA A-fragment (or B-fragment where lane = column and
// "rows" are columns of B, i.e. B^T stored row-major).
// ISA layout: lane l<16 -> row l, K = {k0..k0+7, k0+16..k0+23}
//             lane l>=16 -> row l-16, K = {k0+8..k0+15, k0+24..k0+31}
static __device__ __forceinline__ v16bf load_frag(const bf16* __restrict__ base,
                                                  int ld, int row0, int k0, int lane) {
    int r  = row0 + (lane & 15);
    int hi = (lane >> 4) & 1;
    const bf16* p0 = base + (long)r * ld + k0 + hi * 8;
    v8bf a = *(const v8bf*)p0;
    v8bf b = *(const v8bf*)(p0 + 16);
    return __builtin_shufflevector(a, b, 0,1,2,3,4,5,6,7,8,9,10,11,12,13,14,15);
}

static __device__ __forceinline__ v8f wmma_bf16(v16bf a, v16bf b, v8f c) {
    return __builtin_amdgcn_wmma_f32_16x16x32_bf16(false, a, false, b,
                                                   (short)0, c, false, false);
}

// Async DMA: 16 bytes global -> LDS per lane, tracked by ASYNCcnt (no VGPR round trip).
// GVS mode: global addr = saddr + vaddr32 ; LDS addr = vdst VGPR value.
static __device__ __forceinline__ void async_ld_b128(void* lds_dst, const void* gbase,
                                                     unsigned byte_off) {
    unsigned lds = (unsigned)(size_t)lds_dst;   // low 32 bits of flat LDS addr = LDS offset
    asm volatile("global_load_async_to_lds_b128 %0, %1, %2"
                 :: "v"(lds), "v"(byte_off), "s"(gbase) : "memory");
}
static __device__ __forceinline__ void wait_async() {
    asm volatile("s_wait_asynccnt 0x0" ::: "memory");
}

// ---------- kernel 0a: transpose-convert x [8][256][4096] -> xb_t [8][4096][256] bf16
__global__ void k_cvt_x(const float* __restrict__ x, bf16* __restrict__ xb_t) {
    long i = (long)blockIdx.x * blockDim.x + threadIdx.x;
    if (i >= 8L * 4096 * 256) return;
    int c = (int)(i & 255);
    long t = i >> 8;
    int n = (int)(t & 4095);
    int b = (int)(t >> 12);
    xb_t[i] = (bf16)x[((long)b * 256 + c) * 4096 + n];
}

// ---------- kernel 0b: plain f32 -> bf16 convert (weights)
__global__ void k_cvt_w(const float* __restrict__ w, bf16* __restrict__ o, int n) {
    int i = blockIdx.x * blockDim.x + threadIdx.x;
    if (i < n) o[i] = (bf16)w[i];
}

// ---------- kernel 1: QKV GEMM.  D(4096 tok x 768 ch) = xb_t(4096x256) * Wqkv^T
// Weight tile (64x256 bf16 = 32KB) staged once per block via async global->LDS DMA.
// Epilogue: +bias, scatter into q_t/v_t (token-major bf16) and f_ch (channel-major bf16).
__global__ __launch_bounds__(256) void k_qkv(const bf16* __restrict__ xb_t,
                                             const bf16* __restrict__ wq,
                                             const float* __restrict__ bias,
                                             bf16* __restrict__ q_t,
                                             bf16* __restrict__ v_t,
                                             bf16* __restrict__ f_ch) {
    __shared__ bf16 wlds[64 * 256];
    int b = blockIdx.z, cb = blockIdx.y, tb = blockIdx.x;
    int tid = threadIdx.x, lane = tid & 31, w = tid >> 5;
    const bf16* A = xb_t + (long)b * 4096 * 256;
    const bf16* Wt = wq + (long)cb * 64 * 256;
    // stage weight tile: 2048 x 16B units
    for (int i = tid; i < 2048; i += 256)
        async_ld_b128((char*)wlds + i * 16, Wt, (unsigned)(i * 16));
    int row0 = tb * 128 + w * 16;
    v8f acc[4];
    for (int j = 0; j < 4; ++j) acc[j] = zero_v8f();
    wait_async();
    __syncthreads();
    for (int ks = 0; ks < 8; ++ks) {
        v16bf af = load_frag(A, 256, row0, ks * 32, lane);
        for (int j = 0; j < 4; ++j) {
            v16bf bfr = load_frag(wlds, 256, j * 16, ks * 32, lane);
            acc[j] = wmma_bf16(af, bfr, acc[j]);
        }
    }
    int mb = row0 + ((lane >> 4) << 3);
    for (int j = 0; j < 4; ++j) {
        int o = cb * 64 + j * 16 + (lane & 15);
        float bv = bias[o];
        int t = o >> 8, rem = o & 255, head = rem >> 5, d = rem & 31;
        int bh = b * 8 + head;
        for (int r = 0; r < 8; ++r) {
            float val = acc[j][r] + bv;
            int tok = mb + r;
            if (t == 0)      q_t[((long)bh * 4096 + tok) * 32 + d] = (bf16)val;
            else if (t == 1) v_t[((long)bh * 4096 + tok) * 32 + d] = (bf16)val;
            else             f_ch[((long)bh * 32 + d) * 4096 + tok] = (bf16)val;
        }
    }
}

// ---------- kernel 2a: 8x8 block-mean pooling for centers (from q) and centers_feature (from f)
__global__ void k_pool(const bf16* __restrict__ q_t, const bf16* __restrict__ f_ch,
                       float* __restrict__ centers, float* __restrict__ cfeat) {
    int bh = blockIdx.x, m = blockIdx.y;
    int d = threadIdx.x;            // 32 threads
    int mh = m >> 3, mw = m & 7;
    float sq = 0.f, sf = 0.f;
    for (int r = 0; r < 8; ++r)
        for (int c = 0; c < 8; ++c) {
            int tok = (mh * 8 + r) * 64 + mw * 8 + c;
            sq += (float)q_t[((long)bh * 4096 + tok) * 32 + d];
            sf += (float)f_ch[((long)bh * 32 + d) * 4096 + tok];
        }
    centers[(bh * 64 + m) * 32 + d] = sq * (1.f / 64.f);
    cfeat[(bh * 64 + m) * 32 + d]   = sf * (1.f / 64.f);
}

// ---------- kernel 2b: per-token L2 normalize q -> qn_t bf16 (token-major)
__global__ void k_qnorm(const bf16* __restrict__ q_t, bf16* __restrict__ qn_t) {
    long i = (long)blockIdx.x * blockDim.x + threadIdx.x;
    if (i >= 64L * 4096) return;
    const bf16* p = q_t + i * 32;
    float v[32], s = 0.f;
    for (int d = 0; d < 32; ++d) { v[d] = (float)p[d]; s += v[d] * v[d]; }
    float inv = 1.f / fmaxf(sqrtf(s), 1e-12f);
    bf16* o = qn_t + i * 32;
    for (int d = 0; d < 32; ++d) o[d] = (bf16)(v[d] * inv);
}

// ---------- kernel 3: 3 fused clustering iterations. One block per bh (64 blocks, 8 waves).
// attn tile (64 x 128) never leaves LDS: WMMA -> column softmax -> bf16 -> WMMA accumulate.
// F chunk (32 x 128 bf16 = 8KB) staged per step via async DMA (kills 4x fragment redundancy).
__global__ __launch_bounds__(256) void k_cluster(const float* __restrict__ centers_in,
                                                 const bf16* __restrict__ v_t,
                                                 const bf16* __restrict__ f_ch,
                                                 float* __restrict__ centers_out) {
    __shared__ bf16  cent[64 * 32];
    __shared__ float S[64 * 128];
    __shared__ bf16  A2[64 * 128];
    __shared__ bf16  Fl[32 * 128];
    int bh = blockIdx.x;
    int tid = threadIdx.x, lane = tid & 31, w = tid >> 5;
    const bf16* V = v_t + (long)bh * 4096 * 32;
    const bf16* F = f_ch + (long)bh * 32 * 4096;
    for (int i = tid; i < 2048; i += 256) cent[i] = (bf16)centers_in[bh * 2048 + i];
    __syncthreads();
    int mt = w >> 1, dt = w & 1;
    v8f accC = zero_v8f();
    for (int it = 0; it < 3; ++it) {
        accC = zero_v8f();
        for (int n0 = 0; n0 < 4096; n0 += 128) {
            // async-stage F chunk: rows d=0..31, tokens n0..n0+127 ; 512 x 16B units
            for (int i = tid; i < 512; i += 256) {
                int d = i >> 4, j = i & 15;
                async_ld_b128((char*)Fl + i * 16, F,
                              (unsigned)((d * 4096 + n0 + j * 8) * 2));
            }
            // prefetch next chunk's V stream (consumed exactly once)
            if (n0 + 128 < 4096)
                __builtin_prefetch(V + (long)(n0 + 128) * 32 + tid * 16, 0, 0);
            // S = centers (64x32) x V-chunk^T : wave w owns token subtile w, all 4 M-tiles
            for (int m4 = 0; m4 < 4; ++m4) {
                v16bf a  = load_frag(cent, 32, m4 * 16, 0, lane);
                v16bf bq = load_frag(V, 32, n0 + w * 16, 0, lane);
                v8f s = wmma_bf16(a, bq, zero_v8f());
                int col = w * 16 + (lane & 15);
                int m0  = m4 * 16 + ((lane >> 4) << 3);
                for (int r = 0; r < 8; ++r) S[(m0 + r) * 128 + col] = s[r];
            }
            __syncthreads();
            // column softmax over the 64 centers (axis = M), emit bf16 A2
            if (tid < 128) {
                int col = tid;
                float mx = -1e30f;
                for (int m = 0; m < 64; ++m) mx = fmaxf(mx, S[m * 128 + col]);
                float sum = 0.f;
                for (int m = 0; m < 64; ++m) {
                    float e = __expf(S[m * 128 + col] - mx);
                    S[m * 128 + col] = e; sum += e;
                }
                float inv = 1.f / sum;
                for (int m = 0; m < 64; ++m)
                    A2[m * 128 + col] = (bf16)(S[m * 128 + col] * inv);
            }
            wait_async();
            __syncthreads();
            // centers_new += A2 (64x128) x Fl (128x32); wave w owns tile (mt,dt)
            for (int ks = 0; ks < 4; ++ks) {
                v16bf a  = load_frag(A2, 128, mt * 16, ks * 32, lane);
                v16bf bq = load_frag(Fl, 128, dt * 16, ks * 32, lane);
                accC = wmma_bf16(a, bq, accC);
            }
            __syncthreads();
        }
        // install new centers (bf16) for next iteration
        {
            int col = dt * 16 + (lane & 15);
            int m0  = mt * 16 + ((lane >> 4) << 3);
            for (int r = 0; r < 8; ++r) cent[(m0 + r) * 32 + col] = (bf16)accC[r];
        }
        __syncthreads();
    }
    int col = dt * 16 + (lane & 15);
    int m0  = mt * 16 + ((lane >> 4) << 3);
    for (int r = 0; r < 8; ++r)
        centers_out[bh * 2048 + (m0 + r) * 32 + col] = accC[r];
}

// ---------- kernel 4: cos-sim + sigmoid + hard argmax + out_c. One block per bh.
__global__ __launch_bounds__(256) void k_sim(const float* __restrict__ centers,
                                             const bf16* __restrict__ qn_t,
                                             const bf16* __restrict__ f_ch,
                                             const float* __restrict__ cfeat,
                                             const float* __restrict__ alpha_p,
                                             const float* __restrict__ beta_p,
                                             int* __restrict__ idx_g,
                                             float* __restrict__ simw_g,
                                             float* __restrict__ outc_g) {
    __shared__ bf16  centn[64 * 32];
    __shared__ float S[64 * 128];
    __shared__ bf16  A2[64 * 128];
    __shared__ bf16  Fl[32 * 128];
    __shared__ int   cnt[64];
    int bh = blockIdx.x, tid = threadIdx.x, lane = tid & 31, w = tid >> 5;
    float alpha = alpha_p[0], beta = beta_p[0];
    if (tid < 64) {
        const float* cr = centers + bh * 2048 + tid * 32;
        float s = 0.f;
        for (int d = 0; d < 32; ++d) s += cr[d] * cr[d];
        float inv = 1.f / fmaxf(sqrtf(s), 1e-12f);
        for (int d = 0; d < 32; ++d) centn[tid * 32 + d] = (bf16)(cr[d] * inv);
        cnt[tid] = 0;
    }
    __syncthreads();
    const bf16* Q = qn_t + (long)bh * 4096 * 32;
    const bf16* F = f_ch + (long)bh * 32 * 4096;
    int mt = w >> 1, dt = w & 1;
    v8f accO = zero_v8f();
    for (int n0 = 0; n0 < 4096; n0 += 128) {
        for (int i = tid; i < 512; i += 256) {
            int d = i >> 4, j = i & 15;
            async_ld_b128((char*)Fl + i * 16, F,
                          (unsigned)((d * 4096 + n0 + j * 8) * 2));
        }
        if (n0 + 128 < 4096)
            __builtin_prefetch(Q + (long)(n0 + 128) * 32 + tid * 16, 0, 0);
        for (int m4 = 0; m4 < 4; ++m4) {
            v16bf a  = load_frag(centn, 32, m4 * 16, 0, lane);
            v16bf bq = load_frag(Q, 32, n0 + w * 16, 0, lane);
            v8f s = wmma_bf16(a, bq, zero_v8f());
            int col = w * 16 + (lane & 15);
            int m0  = m4 * 16 + ((lane >> 4) << 3);
            for (int r = 0; r < 8; ++r) S[(m0 + r) * 128 + col] = s[r];
        }
        __syncthreads();
        if (tid < 128) {
            int col = tid;
            float best = -1e30f; int bi = 0;
            for (int m = 0; m < 64; ++m) {
                float sg = 1.f / (1.f + __expf(-(beta + alpha * S[m * 128 + col])));
                S[m * 128 + col] = sg;
                if (sg > best) { best = sg; bi = m; }
            }
            for (int m = 0; m < 64; ++m)
                A2[m * 128 + col] = (m == bi) ? (bf16)best : (bf16)0.f;
            idx_g[bh * 4096 + n0 + col]  = bi;
            simw_g[bh * 4096 + n0 + col] = best;
            atomicAdd(&cnt[bi], 1);
        }
        wait_async();
        __syncthreads();
        for (int ks = 0; ks < 4; ++ks) {
            v16bf a  = load_frag(A2, 128, mt * 16, ks * 32, lane);
            v16bf bq = load_frag(Fl, 128, dt * 16, ks * 32, lane);
            accO = wmma_bf16(a, bq, accO);
        }
        __syncthreads();
    }
    {
        int d  = dt * 16 + (lane & 15);
        int m0 = mt * 16 + ((lane >> 4) << 3);
        for (int r = 0; r < 8; ++r) {
            int m = m0 + r;
            float o = (accO[r] + cfeat[bh * 2048 + m * 32 + d]) / ((float)cnt[m] + 1.f);
            outc_g[bh * 2048 + m * 32 + d] = o;
        }
    }
}

// ---------- kernel 5: one-hot dispatch back to tokens (gather), token-major bf16 output
__global__ void k_dispatch(const float* __restrict__ outc, const int* __restrict__ idx,
                           const float* __restrict__ simw, bf16* __restrict__ out_tok) {
    long i = (long)blockIdx.x * blockDim.x + threadIdx.x;
    if (i >= 64L * 4096) return;
    int bh = (int)(i >> 12), n = (int)(i & 4095);
    int b = bh >> 3, head = bh & 7;
    int m = idx[i];
    float s = simw[i];
    const float* oc = outc + bh * 2048 + m * 32;
    bf16* o = out_tok + ((long)b * 4096 + n) * 256 + head * 32;
    for (int d = 0; d < 32; ++d) o[d] = (bf16)(oc[d] * s);
}

// ---------- kernel 6: proj GEMM + bias, write final f32 [b][o][h][w]
__global__ __launch_bounds__(256) void k_proj(const bf16* __restrict__ out_tok,
                                              const bf16* __restrict__ wp,
                                              const float* __restrict__ bias,
                                              float* __restrict__ out) {
    __shared__ bf16 wlds[64 * 256];
    int b = blockIdx.z, cb = blockIdx.y, tb = blockIdx.x;
    int tid = threadIdx.x, lane = tid & 31, w = tid >> 5;
    const bf16* A = out_tok + (long)b * 4096 * 256;
    const bf16* Wt = wp + (long)cb * 64 * 256;
    for (int i = tid; i < 2048; i += 256)
        async_ld_b128((char*)wlds + i * 16, Wt, (unsigned)(i * 16));
    int row0 = tb * 128 + w * 16;
    v8f acc[4];
    for (int j = 0; j < 4; ++j) acc[j] = zero_v8f();
    wait_async();
    __syncthreads();
    for (int ks = 0; ks < 8; ++ks) {
        v16bf af = load_frag(A, 256, row0, ks * 32, lane);
        for (int j = 0; j < 4; ++j) {
            v16bf bfr = load_frag(wlds, 256, j * 16, ks * 32, lane);
            acc[j] = wmma_bf16(af, bfr, acc[j]);
        }
    }
    int mb = row0 + ((lane >> 4) << 3);
    for (int j = 0; j < 4; ++j) {
        int o = cb * 64 + j * 16 + (lane & 15);
        float bv = bias[o];
        for (int r = 0; r < 8; ++r)
            out[((long)b * 256 + o) * 4096 + mb + r] = acc[j][r] + bv;
    }
}

extern "C" void kernel_launch(void* const* d_in, const int* in_sizes, int n_in,
                              void* d_out, int out_size, void* d_ws, size_t ws_size,
                              hipStream_t stream) {
    const float* x      = (const float*)d_in[0];
    const float* w_qkv  = (const float*)d_in[1];
    const float* b_qkv  = (const float*)d_in[2];
    const float* w_proj = (const float*)d_in[3];
    const float* b_proj = (const float*)d_in[4];
    const float* alpha  = (const float*)d_in[5];
    const float* beta   = (const float*)d_in[6];
    float* out = (float*)d_out;

    char* ws = (char*)d_ws;
    size_t off = 0;
    auto alloc = [&](size_t bytes) -> void* {
        void* p = ws + off;
        off = (off + bytes + 255) & ~(size_t)255;
        return p;
    };
    bf16*  xb_t    = (bf16*)alloc(8L * 4096 * 256 * 2);   // 16 MB
    bf16*  wqkv_b  = (bf16*)alloc(768L * 256 * 2);
    bf16*  wproj_b = (bf16*)alloc(256L * 256 * 2);
    bf16*  q_t     = (bf16*)alloc(64L * 4096 * 32 * 2);   // 16 MB
    bf16*  v_t     = (bf16*)alloc(64L * 4096 * 32 * 2);   // 16 MB
    bf16*  f_ch    = (bf16*)alloc(64L * 32 * 4096 * 2);   // 16 MB
    bf16*  qn_t    = (bf16*)alloc(64L * 4096 * 32 * 2);   // 16 MB
    float* centers = (float*)alloc(64L * 64 * 32 * 4);
    float* cfeat   = (float*)alloc(64L * 64 * 32 * 4);
    float* centers2= (float*)alloc(64L * 64 * 32 * 4);
    float* outc    = (float*)alloc(64L * 64 * 32 * 4);
    int*   idx_g   = (int*)alloc(64L * 4096 * 4);
    float* simw_g  = (float*)alloc(64L * 4096 * 4);
    bf16*  out_tok = (bf16*)alloc(8L * 4096 * 256 * 2);   // 16 MB

    // 0) precision conversion + layout transpose
    k_cvt_x<<<dim3((unsigned)((8L * 4096 * 256 + 255) / 256)), 256, 0, stream>>>(x, xb_t);
    k_cvt_w<<<dim3((768 * 256 + 255) / 256), 256, 0, stream>>>(w_qkv, wqkv_b, 768 * 256);
    k_cvt_w<<<dim3((256 * 256 + 255) / 256), 256, 0, stream>>>(w_proj, wproj_b, 256 * 256);

    // 1) QKV GEMM (WMMA, async-LDS weight staging) with fused bias + q/v/f scatter
    k_qkv<<<dim3(32, 12, 8), 256, 0, stream>>>(xb_t, wqkv_b, b_qkv, q_t, v_t, f_ch);

    // 2) pooled centers / centers_feature, normalized q tokens
    k_pool<<<dim3(64, 64), 32, 0, stream>>>(q_t, f_ch, centers, cfeat);
    k_qnorm<<<dim3((64 * 4096 + 255) / 256), 256, 0, stream>>>(q_t, qn_t);

    // 3) 3 fused clustering iterations (GEMM -> softmax -> GEMM in LDS, async F staging)
    k_cluster<<<dim3(64), 256, 0, stream>>>(centers, v_t, f_ch, centers2);

    // 4) cos-sim + sigmoid + argmax + weighted scatter-combine
    k_sim<<<dim3(64), 256, 0, stream>>>(centers2, qn_t, f_ch, cfeat, alpha, beta,
                                        idx_g, simw_g, outc);

    // 5) dispatch centers back to tokens (one-hot gather)
    k_dispatch<<<dim3((64 * 4096 + 255) / 256), 256, 0, stream>>>(outc, idx_g, simw_g, out_tok);

    // 6) projection GEMM (WMMA, async-LDS weight staging) + bias -> final f32 NCHW output
    k_proj<<<dim3(32, 4, 8), 256, 0, stream>>>(out_tok, wproj_b, b_proj, out);
}
